// Encoder_74388833566985
// MI455X (gfx1250) — matrix operand order repaired and checked by reading, not compiled, not run on hardware
//
#include <hip/hip_runtime.h>
#include <hip/hip_bf16.h>

typedef __attribute__((ext_vector_type(16))) __bf16 v16bf;
typedef __attribute__((ext_vector_type(8)))  __bf16 v8bf;
typedef __attribute__((ext_vector_type(8)))  float  v8f;

#define DEVINL __device__ __forceinline__

static constexpr int N_NODES = 20000;
static constexpr int E_EDGES = 320000;
static constexpr int WPB     = 8;      // waves per block (wave32)
static constexpr int BLOCK   = 32 * WPB;

DEVINL v8f wmma_bf16(v16bf a, v16bf b, v8f c) {
  // v_wmma_f32_16x16x32_bf16 (args: neg_a, A, neg_b, B, c_mod, C, reuse_a, reuse_b)
  return __builtin_amdgcn_wmma_f32_16x16x32_bf16(false, a, false, b, (short)0, c, false, false);
}

// Packed-B layout: element(((c*nt + t)*32 + lane)*16 + j) = B[c*32 + (j>>3)*16 + (lane>>4)*8 + (j&7)][t*16 + (lane&15)]
// Works for both global and LDS-resident packed buffers (inlined -> addrspace inferred).
DEVINL v16bf load_bfrag(const __bf16* pb, int c, int t, int nt, int lane) {
  const __bf16* p = pb + (((c * nt + t) * 32 + lane) << 4);
  v8bf lo = *(const v8bf*)p;
  v8bf hi = *(const v8bf*)(p + 8);
  v16bf r;
#pragma unroll
  for (int j = 0; j < 8; ++j) { r[j] = lo[j]; r[8 + j] = hi[j]; }
  return r;
}

// Cooperative block copy of packed bf16 weights (global -> LDS), 16B granules.
DEVINL void stage_lds(__bf16* s, const __bf16* g, int nelem) {
  const uint4* gs = (const uint4*)g;
  uint4* ss = (uint4*)s;
  for (int i = threadIdx.x; i < (nelem >> 3); i += blockDim.x) ss[i] = gs[i];
}

// ---------------------------------------------------------------------------
// Edge message kernel: one wave = 16 edges.
//   m_s  (16 x MS)  = [xs*sh0 | dot(xv,sh1)]            @ [Wa; Wd]      (K = S_IN+V_IN)
//   m_v_i(16 x MV)  = [xv_i*sh0 | cross_i | xs*sh1_i]   @ [Wc; We; Wb]  (K = 2*V_IN+S_IN)
// inv_sqrt_nn folded into packed weights; results atomically scattered to agg[dst].
// Packed B staged in LDS once per block; fragments served by ds_load_b128.
// ---------------------------------------------------------------------------
template<int S_IN, int V_IN, int MS, int MV>
__global__ void edge_msg_kernel(const float* __restrict__ xs_g, int xs_str,
                                const float* __restrict__ xv_g, int xv_str,
                                const float* __restrict__ eattr,
                                const int* __restrict__ src,
                                const int* __restrict__ dst,
                                const __bf16* __restrict__ pbS,
                                const __bf16* __restrict__ pbV,
                                float* __restrict__ agg_s,
                                float* __restrict__ agg_v,
                                int nedges)
{
  constexpr int KS  = S_IN + V_IN;
  constexpr int KV  = 2 * V_IN + S_IN;
  constexpr int NTS = MS / 16;
  constexpr int NTV = MV / 16;
  constexpr int NBS = KS * MS;
  constexpr int NBV = KV * MV;

  __shared__ __align__(16) __bf16 sB[NBS + NBV];
  stage_lds(sB, pbS, NBS);
  stage_lds(sB + NBS, pbV, NBV);
  __syncthreads();

  const int lane = threadIdx.x & 31;
  const int tile = blockIdx.x * WPB + (threadIdx.x >> 5);
  const int base = tile * 16;
  if (base >= nedges) return;
  const int h = lane >> 4;
  const int m = lane & 15;
  const int e = base + m;
  const int sN = src[e];
  const float sh0  = eattr[4 * e + 0];
  const float sh1x = eattr[4 * e + 1];
  const float sh1y = eattr[4 * e + 2];
  const float sh1z = eattr[4 * e + 3];
  const float sh1[3] = {sh1x, sh1y, sh1z};
  const float* xsrow = xs_g + (size_t)sN * xs_str;
  const float* xvrow = xv_g + (size_t)sN * xv_str;

  int drow[8];
#pragma unroll
  for (int r = 0; r < 8; ++r) drow[r] = dst[base + r + 8 * h];

  // ---- scalar messages ----
  {
    v8f acc[NTS] = {};
    for (int c = 0; c < KS / 32; ++c) {
      v16bf a;
#pragma unroll
      for (int r2 = 0; r2 < 2; ++r2) {
        const int k0 = c * 32 + r2 * 16 + h * 8;
        if (k0 < S_IN) {
          const float* p = xsrow + k0;
#pragma unroll
          for (int j = 0; j < 8; ++j) a[r2 * 8 + j] = (__bf16)(p[j] * sh0);
        } else {
          const float* q = xvrow + (size_t)(k0 - S_IN) * 3;
#pragma unroll
          for (int j = 0; j < 8; ++j)
            a[r2 * 8 + j] = (__bf16)(q[3 * j] * sh1x + q[3 * j + 1] * sh1y + q[3 * j + 2] * sh1z);
        }
      }
#pragma unroll
      for (int t = 0; t < NTS; ++t)
        acc[t] = wmma_bf16(a, load_bfrag(sB, c, t, NTS, lane), acc[t]);
    }
#pragma unroll
    for (int t = 0; t < NTS; ++t) {
      const int ch = t * 16 + m;
#pragma unroll
      for (int r = 0; r < 8; ++r)
        atomicAdd(&agg_s[(size_t)drow[r] * MS + ch], acc[t][r]);
    }
  }

  // ---- vector messages (3 coordinate GEMMs share B fragments) ----
  {
    v8f acc[3][NTV] = {};
    for (int c = 0; c < KV / 32; ++c) {
      v16bf av[3];
#pragma unroll
      for (int i = 0; i < 3; ++i) {
        const int i1 = (i + 1) % 3, i2 = (i + 2) % 3;
#pragma unroll
        for (int r2 = 0; r2 < 2; ++r2) {
          const int k0 = c * 32 + r2 * 16 + h * 8;
          if (k0 < V_IN) {
            const float* q = xvrow + (size_t)k0 * 3;
#pragma unroll
            for (int j = 0; j < 8; ++j) av[i][r2 * 8 + j] = (__bf16)(q[3 * j + i] * sh0);
          } else if (k0 < 2 * V_IN) {
            const float* q = xvrow + (size_t)(k0 - V_IN) * 3;
#pragma unroll
            for (int j = 0; j < 8; ++j)
              av[i][r2 * 8 + j] = (__bf16)(q[3 * j + i1] * sh1[i2] - q[3 * j + i2] * sh1[i1]);
          } else {
            const float* p = xsrow + (k0 - 2 * V_IN);
#pragma unroll
            for (int j = 0; j < 8; ++j) av[i][r2 * 8 + j] = (__bf16)(p[j] * sh1[i]);
          }
        }
      }
#pragma unroll
      for (int t = 0; t < NTV; ++t) {
        v16bf b = load_bfrag(sB + NBS, c, t, NTV, lane);
#pragma unroll
        for (int i = 0; i < 3; ++i) acc[i][t] = wmma_bf16(av[i], b, acc[i][t]);
      }
    }
#pragma unroll
    for (int t = 0; t < NTV; ++t) {
      const int ch = t * 16 + m;
#pragma unroll
      for (int r = 0; r < 8; ++r) {
        const size_t o = ((size_t)drow[r] * MV + ch) * 3;
#pragma unroll
        for (int i = 0; i < 3; ++i) atomicAdd(&agg_v[o + i], acc[i][t][r]);
      }
    }
  }
}

// ---------------------------------------------------------------------------
// Node scalar update: s = [agg_s | x_s] @ [lin_s; skip_s] + emb_z[z]
//   x_s_out = silu(s[:S_OUT]);  gate = sigmoid(s[S_OUT:])
// ---------------------------------------------------------------------------
template<int MS, int S_IN, int SV, int S_OUT>
__global__ void node_s_kernel(const float* __restrict__ agg_s,
                              const float* __restrict__ xs_g, int xs_str,
                              const __bf16* __restrict__ pb,
                              const float* __restrict__ embz,
                              const int* __restrict__ z,
                              float* __restrict__ xs_out, int xo_str,
                              float* __restrict__ gate,
                              int nnodes)
{
  constexpr int K = MS + S_IN;
  constexpr int NT = SV / 16;
  constexpr int V_OUT = SV - S_OUT;
  constexpr int NB = K * SV;

  __shared__ __align__(16) __bf16 sB[NB];
  stage_lds(sB, pb, NB);
  __syncthreads();

  const int lane = threadIdx.x & 31;
  const int tile = blockIdx.x * WPB + (threadIdx.x >> 5);
  const int base = tile * 16;
  if (base >= nnodes) return;
  const int h = lane >> 4;
  const int m = lane & 15;
  const int nd = base + m;
  const float* arow = agg_s + (size_t)nd * MS;
  const float* xrow = xs_g + (size_t)nd * xs_str;

  v8f acc[NT] = {};
  for (int c = 0; c < K / 32; ++c) {
    v16bf a;
#pragma unroll
    for (int r2 = 0; r2 < 2; ++r2) {
      const int k0 = c * 32 + r2 * 16 + h * 8;
      const float* p = (k0 < MS) ? (arow + k0) : (xrow + (k0 - MS));
#pragma unroll
      for (int j = 0; j < 8; ++j) a[r2 * 8 + j] = (__bf16)p[j];
    }
#pragma unroll
    for (int t = 0; t < NT; ++t)
      acc[t] = wmma_bf16(a, load_bfrag(sB, c, t, NT, lane), acc[t]);
  }

  int zr[8];
#pragma unroll
  for (int r = 0; r < 8; ++r) zr[r] = z[base + r + 8 * h];

#pragma unroll
  for (int t = 0; t < NT; ++t) {
    const int ch = t * 16 + m;
#pragma unroll
    for (int r = 0; r < 8; ++r) {
      const int nrow = base + r + 8 * h;
      const float sv = acc[t][r] + embz[(size_t)zr[r] * SV + ch];
      const float sig = 1.0f / (1.0f + __expf(-sv));
      if (ch < S_OUT) xs_out[(size_t)nrow * xo_str + ch] = sv * sig;
      else            gate[(size_t)nrow * V_OUT + (ch - S_OUT)] = sig;
    }
  }
}

// ---------------------------------------------------------------------------
// Node vector update: v_i = [agg_v_i | x_v_i] @ [lin_v; skip_v];  out = v * gate
// K padded with zeros up to multiple of 32.
// ---------------------------------------------------------------------------
template<int MV, int V_IN, int V_OUT>
__global__ void node_v_kernel(const float* __restrict__ agg_v,
                              const float* __restrict__ xv_g, int xv_str,
                              const __bf16* __restrict__ pb,
                              const float* __restrict__ gate,
                              float* __restrict__ xv_out, int xo_str,
                              int nnodes)
{
  constexpr int KR = MV + V_IN;
  constexpr int KC = (KR + 31) / 32;
  constexpr int NT = V_OUT / 16;
  constexpr int NB = KC * 32 * V_OUT;

  __shared__ __align__(16) __bf16 sB[NB];
  stage_lds(sB, pb, NB);
  __syncthreads();

  const int lane = threadIdx.x & 31;
  const int tile = blockIdx.x * WPB + (threadIdx.x >> 5);
  const int base = tile * 16;
  if (base >= nnodes) return;
  const int h = lane >> 4;
  const int m = lane & 15;
  const int nd = base + m;
  const float* arow = agg_v + (size_t)nd * MV * 3;
  const float* xrow = xv_g + (size_t)nd * xv_str;

  v8f acc[3][NT] = {};
  for (int c = 0; c < KC; ++c) {
    v16bf av[3];
#pragma unroll
    for (int i = 0; i < 3; ++i) {
#pragma unroll
      for (int r2 = 0; r2 < 2; ++r2) {
        const int k0 = c * 32 + r2 * 16 + h * 8;
        if (k0 < MV) {
          const float* q = arow + (size_t)k0 * 3;
#pragma unroll
          for (int j = 0; j < 8; ++j) av[i][r2 * 8 + j] = (__bf16)q[3 * j + i];
        } else if (k0 < KR) {
          const float* q = xrow + (size_t)(k0 - MV) * 3;
#pragma unroll
          for (int j = 0; j < 8; ++j) av[i][r2 * 8 + j] = (__bf16)q[3 * j + i];
        } else {
#pragma unroll
          for (int j = 0; j < 8; ++j) av[i][r2 * 8 + j] = (__bf16)0.0f;
        }
      }
    }
#pragma unroll
    for (int t = 0; t < NT; ++t) {
      v16bf b = load_bfrag(sB, c, t, NT, lane);
#pragma unroll
      for (int i = 0; i < 3; ++i) acc[i][t] = wmma_bf16(av[i], b, acc[i][t]);
    }
  }

#pragma unroll
  for (int t = 0; t < NT; ++t) {
    const int ch = t * 16 + m;
#pragma unroll
    for (int r = 0; r < 8; ++r) {
      const int nrow = base + r + 8 * h;
      const float g = gate[(size_t)nrow * V_OUT + ch];
#pragma unroll
      for (int i = 0; i < 3; ++i)
        xv_out[(size_t)nrow * xo_str + ch * 3 + i] = acc[i][t][r] * g;
    }
  }
}

// Pack up to 3 stacked row-major f32 weight blocks into fragment-native bf16 layout.
__global__ void pack_b_kernel(const float* __restrict__ W0, int r0,
                              const float* __restrict__ W1, int r1,
                              const float* __restrict__ W2, int r2,
                              int ncols, float scale,
                              __bf16* __restrict__ out, int total)
{
  const int nt = ncols >> 4;
  for (int idx = blockIdx.x * blockDim.x + threadIdx.x; idx < total;
       idx += gridDim.x * blockDim.x) {
    const int j = idx & 15;
    const int l = (idx >> 4) & 31;
    const int rest = idx >> 9;
    const int t = rest % nt;
    const int c = rest / nt;
    const int k = c * 32 + (j >> 3) * 16 + (l >> 4) * 8 + (j & 7);
    const int col = t * 16 + (l & 15);
    float v = 0.0f;
    if (k < r0)                 v = W0[(size_t)k * ncols + col];
    else if (k < r0 + r1)       v = W1[(size_t)(k - r0) * ncols + col];
    else if (k < r0 + r1 + r2)  v = W2[(size_t)(k - r0 - r1) * ncols + col];
    out[idx] = (__bf16)(v * scale);
  }
}

__global__ void zero_kernel(float* __restrict__ p, int n) {
  int i = blockIdx.x * blockDim.x + threadIdx.x;
  if (i < n) p[i] = 0.0f;
}

extern "C" void kernel_launch(void* const* d_in, const int* in_sizes, int n_in,
                              void* d_out, int out_size, void* d_ws, size_t ws_size,
                              hipStream_t stream)
{
  (void)in_sizes; (void)n_in; (void)out_size; (void)ws_size;
  const int N = N_NODES, E = E_EDGES;
  const float inv_sqrt_nn = 0.25f;  // 1/sqrt(16), folded into edge weights

  const float* x0    = (const float*)d_in[0];
  const float* eattr = (const float*)d_in[1];
  // d_in[2] = canonical (unused by the reference forward pass)
  const float* P[3][10];
  for (int l = 0; l < 3; ++l)
    for (int k = 0; k < 10; ++k)
      P[l][k] = (const float*)d_in[3 + 10 * l + k];
  // param order (dict insertion): 0:Wa 1:Wd 2:Wb 3:Wc 4:We 5:lin_s 6:lin_v 7:skip_s 8:skip_v 9:emb_z
  const int* eidx = (const int*)d_in[33];
  const int* src = eidx;
  const int* dst = eidx + E;
  const int* z = (const int*)d_in[34];

  char* w = (char*)d_ws;
  auto alloc = [&](size_t bytes) -> void* {
    void* p = (void*)w;
    w += (bytes + 255) & ~(size_t)255;
    return p;
  };
  float* agg_s = (float*)alloc((size_t)N * 64 * 4);
  float* agg_v = (float*)alloc((size_t)N * 96 * 4);
  float* xs_a  = (float*)alloc((size_t)N * 64 * 4);
  float* xv_a  = (float*)alloc((size_t)N * 96 * 4);
  float* xs_b  = (float*)alloc((size_t)N * 128 * 4);
  float* xv_b  = (float*)alloc((size_t)N * 192 * 4);
  float* gate  = (float*)alloc((size_t)N * 64 * 4);

  const int S_IN[3] = {64, 64, 128}, V_IN[3] = {32, 32, 64};
  const int MSd[3]  = {32, 64, 64},  MVd[3]  = {16, 32, 32};
  const int SVd[3]  = {96, 192, 192}, VOUT[3] = {32, 64, 64};

  __bf16* pbSe[3]; __bf16* pbVe[3]; __bf16* pbSn[3]; __bf16* pbVn[3];
  for (int l = 0; l < 3; ++l) {
    const int KS  = S_IN[l] + V_IN[l];
    const int KV  = 2 * V_IN[l] + S_IN[l];
    const int KN  = MSd[l] + S_IN[l];
    const int KVN = ((MVd[l] + V_IN[l] + 31) / 32) * 32;
    pbSe[l] = (__bf16*)alloc((size_t)KS * MSd[l] * 2);
    pbVe[l] = (__bf16*)alloc((size_t)KV * MVd[l] * 2);
    pbSn[l] = (__bf16*)alloc((size_t)KN * SVd[l] * 2);
    pbVn[l] = (__bf16*)alloc((size_t)KVN * VOUT[l] * 2);
  }

  auto pack = [&](const float* W0, int r0, const float* W1, int r1,
                  const float* W2, int r2, int ncols, int kpad, float scale, __bf16* out) {
    const int total = kpad * ncols;
    pack_b_kernel<<<(total + 255) / 256, 256, 0, stream>>>(W0, r0, W1, r1, W2, r2,
                                                           ncols, scale, out, total);
  };
  auto zero = [&](float* p, int n) {
    zero_kernel<<<(n + 255) / 256, 256, 0, stream>>>(p, n);
  };

  for (int l = 0; l < 3; ++l) {
    pack(P[l][0], S_IN[l], P[l][1], V_IN[l], nullptr, 0,
         MSd[l], S_IN[l] + V_IN[l], inv_sqrt_nn, pbSe[l]);
    pack(P[l][3], V_IN[l], P[l][4], V_IN[l], P[l][2], S_IN[l],
         MVd[l], 2 * V_IN[l] + S_IN[l], inv_sqrt_nn, pbVe[l]);
    pack(P[l][5], MSd[l], P[l][7], S_IN[l], nullptr, 0,
         SVd[l], MSd[l] + S_IN[l], 1.0f, pbSn[l]);
    pack(P[l][6], MVd[l], P[l][8], V_IN[l], nullptr, 0,
         VOUT[l], ((MVd[l] + V_IN[l] + 31) / 32) * 32, 1.0f, pbVn[l]);
  }

  const int ET  = E / 16;          // 20000 edge tiles
  const int NTL = (N + 15) / 16;   // 1250 node tiles
  const dim3 blk(BLOCK);           // 8 waves per block (wave32)
  const int ebl = (ET + WPB - 1) / WPB;
  const int nbl = (NTL + WPB - 1) / WPB;

  float* out_f = (float*)d_out;

  // ---- Layer 1: (64,32) -> (32,16) -> (64,32) ----
  zero(agg_s, N * 32); zero(agg_v, N * 16 * 3);
  edge_msg_kernel<64, 32, 32, 16><<<ebl, blk, 0, stream>>>(
      x0, 160, x0 + 64, 160, eattr, src, dst, pbSe[0], pbVe[0], agg_s, agg_v, E);
  node_s_kernel<32, 64, 96, 64><<<nbl, blk, 0, stream>>>(
      agg_s, x0, 160, pbSn[0], P[0][9], z, xs_a, 64, gate, N);
  node_v_kernel<16, 32, 32><<<nbl, blk, 0, stream>>>(
      agg_v, x0 + 64, 160, pbVn[0], gate, xv_a, 96, N);

  // ---- Layer 2: (64,32) -> (64,32) -> (128,64) ----
  zero(agg_s, N * 64); zero(agg_v, N * 32 * 3);
  edge_msg_kernel<64, 32, 64, 32><<<ebl, blk, 0, stream>>>(
      xs_a, 64, xv_a, 96, eattr, src, dst, pbSe[1], pbVe[1], agg_s, agg_v, E);
  node_s_kernel<64, 64, 192, 128><<<nbl, blk, 0, stream>>>(
      agg_s, xs_a, 64, pbSn[1], P[1][9], z, xs_b, 128, gate, N);
  node_v_kernel<32, 32, 64><<<nbl, blk, 0, stream>>>(
      agg_v, xv_a, 96, pbVn[1], gate, xv_b, 192, N);

  // ---- Layer 3: (128,64) -> (64,32) -> (128,64), write straight to d_out ----
  zero(agg_s, N * 64); zero(agg_v, N * 32 * 3);
  edge_msg_kernel<128, 64, 64, 32><<<ebl, blk, 0, stream>>>(
      xs_b, 128, xv_b, 192, eattr, src, dst, pbSe[2], pbVe[2], agg_s, agg_v, E);
  node_s_kernel<64, 128, 192, 128><<<nbl, blk, 0, stream>>>(
      agg_s, xs_b, 128, pbSn[2], P[2][9], z, out_f, 320, gate, N);
  node_v_kernel<32, 64, 64><<<nbl, blk, 0, stream>>>(
      agg_v, xv_b, 192, pbVn[2], gate, out_f + 128, 320, N);
}
